// MultiModalMultiHeadAggregatedAttention_70196945486154
// MI455X (gfx1250) — matrix operand order repaired and checked by reading, not compiled, not run on hardware
//
#include <hip/hip_runtime.h>
#include <hip/hip_bf16.h>

typedef __bf16 bf16_t;
typedef __attribute__((ext_vector_type(16))) __bf16 v16bf;
typedef __attribute__((ext_vector_type(8)))  __bf16 v8bf;
typedef __attribute__((ext_vector_type(4)))  __bf16 v4bf;
typedef __attribute__((ext_vector_type(8)))  float  v8f;

#define B_   2
#define S_   2048
#define HID_ 1024
#define NH_  16
#define HD_  64
#define M_   (B_*S_)   // 4096

static __device__ __forceinline__ v16bf cat16(v8bf lo, v8bf hi) {
  return __builtin_shufflevector(lo, hi, 0,1,2,3,4,5,6,7,8,9,10,11,12,13,14,15);
}

static __device__ __forceinline__ v8f wmma_bf16(v16bf a, v16bf b, v8f c) {
  // D = A(16x32) * B(32x16) + C, f32 accumulate
  return __builtin_amdgcn_wmma_f32_16x16x32_bf16(false, a, false, b, (short)0, c, false, false);
}

// ---------------------------------------------------------------------------
// f32 -> bf16 conversion (vectorized x4)
// ---------------------------------------------------------------------------
__global__ void cvt_f32_to_bf16(const float* __restrict__ in,
                                bf16_t* __restrict__ out, int n) {
  int i = (blockIdx.x * blockDim.x + threadIdx.x) * 4;
  if (i >= n) return;
  float4 v = *(const float4*)(in + i);
  v4bf o;
  o[0] = (bf16_t)v.x; o[1] = (bf16_t)v.y; o[2] = (bf16_t)v.z; o[3] = (bf16_t)v.w;
  *(v4bf*)(out + i) = o;
}

// ---------------------------------------------------------------------------
// NT GEMM with WMMA bf16:  D(M,N) = A(MxK) * W(NxK)^T + bias
// Block: 256 thr = 8 waves arranged 4(M) x 2(N); block tile 128x128,
// wave tile 32x64 (2 A-frags x 4 B-frags -> 8 WMMA per 32-K step; each
// B-frag reused by two WMMAs, each A-frag by four).
// mode 0: Q  -> *logn(head), bf16 (B,H,S,D)
// mode 1: K  ->              bf16 (B,H,S,D)
// mode 2: V  ->              bf16 (B,H,D,S)   (transposed for PV WMMA)
// mode 3: final -> f32 row-major (M,N)
// ---------------------------------------------------------------------------
__global__ __launch_bounds__(256) void gemm_nt_bf16(
    const bf16_t* __restrict__ A, const bf16_t* __restrict__ W,
    const float* __restrict__ bias, void* __restrict__ outp,
    int N, int K, int mode)
{
  const int lane = threadIdx.x & 31;
  const int wv   = threadIdx.x >> 5;     // 0..7
  const int wn   = wv & 1;               // 2 waves across N
  const int wm   = wv >> 1;              // 4 waves across M
  const int nl   = lane & 15;
  const int g    = lane >> 4;
  const int m0   = blockIdx.y * 128 + wm * 32;
  const int n0   = blockIdx.x * 128 + wn * 64;
  const int ka   = g ? 8 : 0;    // A-frag K sub-chunk per ISA 16-bit A layout
  const int kb   = g ? 8 : 0;    // B-frag K sub-chunk, same order as A-frag

  const bf16_t* arow0 = A + (size_t)(m0 + nl) * K;
  const bf16_t* arow1 = A + (size_t)(m0 + 16 + nl) * K;
  const bf16_t* bp[4];
#pragma unroll
  for (int t = 0; t < 4; ++t)
    bp[t] = W + (size_t)(n0 + t*16 + nl) * K + kb;

  v8f acc[2][4];
#pragma unroll
  for (int mi = 0; mi < 2; ++mi)
#pragma unroll
    for (int t = 0; t < 4; ++t)
#pragma unroll
      for (int i = 0; i < 8; ++i) acc[mi][t][i] = 0.f;

  for (int k = 0; k < K; k += 32) {
    __builtin_prefetch(arow0 + k + 512, 0, 3);   // WGP-scope: pull to near cache
    __builtin_prefetch(arow1 + k + 512, 0, 3);
    v16bf af0 = cat16(*(const v8bf*)(arow0 + k + ka),
                      *(const v8bf*)(arow0 + k + ka + 16));
    v16bf af1 = cat16(*(const v8bf*)(arow1 + k + ka),
                      *(const v8bf*)(arow1 + k + ka + 16));
#pragma unroll
    for (int t = 0; t < 4; ++t) {
      v16bf wf = cat16(*(const v8bf*)(bp[t] + k),
                       *(const v8bf*)(bp[t] + k + 16));
      acc[0][t] = wmma_bf16(af0, wf, acc[0][t]);
      acc[1][t] = wmma_bf16(af1, wf, acc[1][t]);
    }
  }

#pragma unroll
  for (int mi = 0; mi < 2; ++mi) {
#pragma unroll
    for (int t = 0; t < 4; ++t) {
      const int n = n0 + t*16 + nl;
      const float bi = bias[n];
      float lnsc = 1.f;
      if (mode == 0) {
        const int h = n >> 6;
        const float l = __logf((float)(65 + h)) * (1.0f / __logf(2048.0f));
        lnsc = fmaxf(l, 1.0f);
      }
#pragma unroll
      for (int i = 0; i < 8; ++i) {
        const int m = m0 + mi*16 + i + g*8;   // C-frag row mapping
        float v = acc[mi][t][i] + bi;
        if (mode == 3) {
          ((float*)outp)[(size_t)m * N + n] = v;
        } else {
          const int b = m >> 11, s = m & (S_-1);
          const int h = n >> 6,  d = n & (HD_-1);
          if (mode == 0) v *= lnsc;
          size_t idx;
          if (mode == 2) idx = (((size_t)b*NH_ + h)*HD_ + d)*S_ + s;   // (B,H,D,S)
          else           idx = (((size_t)b*NH_ + h)*S_ + s)*HD_ + d;   // (B,H,S,D)
          ((bf16_t*)outp)[idx] = (bf16_t)v;
        }
      }
    }
  }
}

// ---------------------------------------------------------------------------
// Causal flash attention, one (branch, batch*head, 64-row q tile) per block.
// 4 waves/block; each wave: 16 q rows x D=64, looping 32-wide K tiles.
// Writes att[(b*S+q)][branch*HID + h*HD + d] * exp(agg_w[col]) as bf16.
// ---------------------------------------------------------------------------
__global__ __launch_bounds__(128) void flash_attn_kernel(
    const bf16_t* __restrict__ Qg, const bf16_t* __restrict__ Kg,
    const bf16_t* __restrict__ Vtg, const float* __restrict__ aggw,
    bf16_t* __restrict__ att)
{
  __shared__ __align__(16) bf16_t pbuf[4][16*32];   // per-wave P staging
  const int lane = threadIdx.x & 31;
  const int wv   = threadIdx.x >> 5;
  const int nl   = lane & 15;
  const int g    = lane >> 4;
  const int ka   = g ? 8 : 0;
  const int kb   = g ? 8 : 0;
  const int branch = blockIdx.z;
  const int bh     = blockIdx.y;               // b*NH + h
  const int q0     = blockIdx.x * 64 + wv * 16;

  const size_t hoff = ((size_t)branch * (B_*NH_) + bh) * (size_t)(S_*HD_);
  const bf16_t* Qh = Qg  + hoff;
  const bf16_t* Kh = Kg  + hoff;
  const bf16_t* Vh = Vtg + hoff;               // (D,S) per head

  // Q fragments (16x32 for d=0..31 and d=32..63), kept in registers
  const bf16_t* qrow = Qh + (size_t)(q0 + nl) * HD_;
  v16bf qa[2];
  qa[0] = cat16(*(const v8bf*)(qrow + ka),      *(const v8bf*)(qrow + ka + 16));
  qa[1] = cat16(*(const v8bf*)(qrow + 32 + ka), *(const v8bf*)(qrow + 32 + ka + 16));

  v8f o[4];
#pragma unroll
  for (int t = 0; t < 4; ++t)
#pragma unroll
    for (int i = 0; i < 8; ++i) o[t][i] = 0.f;
  float mrow[8], lrow[8];
#pragma unroll
  for (int i = 0; i < 8; ++i) { mrow[i] = -__builtin_inff(); lrow[i] = 0.f; }

  const int nkt = (q0 + 16 + 31) >> 5;         // causal trip count for this wave
  for (int kt = 0; kt < nkt; ++kt) {
    const int k0 = kt << 5;

    // --- scores: S(16x32) = Q(16x64) * K(32x64)^T ---
    v8f sc[2];
#pragma unroll
    for (int t = 0; t < 2; ++t)
#pragma unroll
      for (int i = 0; i < 8; ++i) sc[t][i] = 0.f;
#pragma unroll
    for (int t = 0; t < 2; ++t) {
      const bf16_t* kcol = Kh + (size_t)(k0 + t*16 + nl)*HD_ + kb;
      __builtin_prefetch(kcol + 32*HD_, 0, 3);   // next K tile
#pragma unroll
      for (int ds = 0; ds < 2; ++ds) {
        v16bf kf = cat16(*(const v8bf*)(kcol + ds*32),
                         *(const v8bf*)(kcol + ds*32 + 16));
        sc[t] = wmma_bf16(qa[ds], kf, sc[t]);
      }
    }

    // --- online softmax (f32) ---
    float p0a[8], p1a[8];
#pragma unroll
    for (int i = 0; i < 8; ++i) {
      const int qg_ = q0 + i + g*8;
      float s0 = sc[0][i] * 0.125f;            // 1/sqrt(64)
      float s1 = sc[1][i] * 0.125f;
      if (k0 + nl      > qg_) s0 = -__builtin_inff();
      if (k0 + 16 + nl > qg_) s1 = -__builtin_inff();
      float tm = fmaxf(s0, s1);
      tm = fmaxf(tm, __shfl_xor(tm, 1, 32));
      tm = fmaxf(tm, __shfl_xor(tm, 2, 32));
      tm = fmaxf(tm, __shfl_xor(tm, 4, 32));
      tm = fmaxf(tm, __shfl_xor(tm, 8, 32));
      const float mo  = mrow[i];
      const float mn  = fmaxf(mo, tm);
      const float scl = __expf(mo - mn);
      const float p0  = __expf(s0 - mn);
      const float p1  = __expf(s1 - mn);
      float rs = p0 + p1;
      rs += __shfl_xor(rs, 1, 32);
      rs += __shfl_xor(rs, 2, 32);
      rs += __shfl_xor(rs, 4, 32);
      rs += __shfl_xor(rs, 8, 32);
      lrow[i] = lrow[i]*scl + rs;
      mrow[i] = mn;
#pragma unroll
      for (int t4 = 0; t4 < 4; ++t4) o[t4][i] *= scl;
      p0a[i] = p0; p1a[i] = p1;
    }

    // --- stage P (16x32 bf16) through per-wave LDS: C-layout -> A-layout ---
    bf16_t* pr = &pbuf[wv][0];
#pragma unroll
    for (int i = 0; i < 8; ++i) {
      const int r = i + g*8;
      pr[r*32 + nl]      = (bf16_t)p0a[i];
      pr[r*32 + 16 + nl] = (bf16_t)p1a[i];
    }
    asm volatile("s_wait_dscnt 0x0" ::: "memory");   // same-wave DS in order
    const bf16_t* prow = &pbuf[wv][nl*32 + ka];
    v16bf pa = cat16(*(const v8bf*)(prow), *(const v8bf*)(prow + 16));

    // --- O(16x64) += P(16x32) * V(32x64), V stored transposed (D,S) ---
#pragma unroll
    for (int dt = 0; dt < 4; ++dt) {
      const bf16_t* vcol = Vh + (size_t)(dt*16 + nl)*S_ + k0 + kb;
      __builtin_prefetch(vcol + 32, 0, 3);       // next V tile
      v16bf vf = cat16(*(const v8bf*)(vcol), *(const v8bf*)(vcol + 16));
      o[dt] = wmma_bf16(pa, vf, o[dt]);
    }
  }

  // --- epilogue: normalize, fold in exp(agg_w), store bf16 into concat att ---
  const int h = bh & (NH_-1);
  const int b = bh >> 4;
#pragma unroll
  for (int dt = 0; dt < 4; ++dt) {
    const int d   = dt*16 + nl;
    const int col = branch*HID_ + h*HD_ + d;
    const float asc = __expf(aggw[col]);
#pragma unroll
    for (int i = 0; i < 8; ++i) {
      const int qg_ = q0 + i + g*8;
      const size_t m = (size_t)b * S_ + qg_;
      const float val = (o[dt][i] / lrow[i]) * asc;
      att[m * (size_t)(2*HID_) + col] = (bf16_t)val;
    }
  }
}

// ---------------------------------------------------------------------------
extern "C" void kernel_launch(void* const* d_in, const int* in_sizes, int n_in,
                              void* d_out, int out_size, void* d_ws, size_t ws_size,
                              hipStream_t stream)
{
  (void)in_sizes; (void)n_in; (void)out_size; (void)ws_size;
  const float* eeg   = (const float*)d_in[0];
  const float* ecg   = (const float*)d_in[1];
  const float* Wq[2] = {(const float*)d_in[2],  (const float*)d_in[8]};
  const float* bq[2] = {(const float*)d_in[3],  (const float*)d_in[9]};
  const float* Wk[2] = {(const float*)d_in[4],  (const float*)d_in[10]};
  const float* bk[2] = {(const float*)d_in[5],  (const float*)d_in[11]};
  const float* Wv[2] = {(const float*)d_in[6],  (const float*)d_in[12]};
  const float* bv[2] = {(const float*)d_in[7],  (const float*)d_in[13]};
  const float* aggw  = (const float*)d_in[14];
  const float* Wout  = (const float*)d_in[15];
  const float* bout  = (const float*)d_in[16];

  bf16_t* ws = (bf16_t*)d_ws;
  const size_t XSZ = (size_t)M_*HID_;        // 4M elems
  const size_t WSZ = (size_t)HID_*HID_;      // 1M elems
  const size_t QSZ = (size_t)B_*NH_*S_*HD_;  // 4M elems
  size_t off = 0;
  bf16_t* x16   = ws + off; off += 2*XSZ;                 // inputs (branch-major)
  bf16_t* w16   = ws + off; off += 6*WSZ;                 // [br*3 + {q,k,v}]
  bf16_t* wo16  = ws + off; off += (size_t)HID_*2*HID_;   // W_out
  bf16_t* Q16   = ws + off; off += 2*QSZ;                 // (br,B,H,S,D)
  bf16_t* K16   = ws + off; off += 2*QSZ;                 // (br,B,H,S,D)
  bf16_t* Vt16  = ws + off; off += 2*QSZ;                 // (br,B,H,D,S)
  bf16_t* att16 = ws + off; off += (size_t)M_*2*HID_;     // concat, scaled

  auto cvt = [&](const float* src, bf16_t* dst, size_t n) {
    cvt_f32_to_bf16<<<dim3((unsigned)(n/1024)), dim3(256), 0, stream>>>(src, dst, (int)n);
  };
  cvt(eeg, x16,        XSZ);
  cvt(ecg, x16 + XSZ,  XSZ);
  for (int br = 0; br < 2; ++br) {
    cvt(Wq[br], w16 + (size_t)(br*3+0)*WSZ, WSZ);
    cvt(Wk[br], w16 + (size_t)(br*3+1)*WSZ, WSZ);
    cvt(Wv[br], w16 + (size_t)(br*3+2)*WSZ, WSZ);
  }
  cvt(Wout, wo16, (size_t)HID_*2*HID_);

  dim3 gproj(HID_/128, M_/128);   // (8, 32), 128x128 block tiles
  for (int br = 0; br < 2; ++br) {
    gemm_nt_bf16<<<gproj, 256, 0, stream>>>(x16 + br*XSZ, w16 + (size_t)(br*3+0)*WSZ,
                                            bq[br], Q16  + br*QSZ, HID_, HID_, 0);
    gemm_nt_bf16<<<gproj, 256, 0, stream>>>(x16 + br*XSZ, w16 + (size_t)(br*3+1)*WSZ,
                                            bk[br], K16  + br*QSZ, HID_, HID_, 1);
    gemm_nt_bf16<<<gproj, 256, 0, stream>>>(x16 + br*XSZ, w16 + (size_t)(br*3+2)*WSZ,
                                            bv[br], Vt16 + br*QSZ, HID_, HID_, 2);
  }

  flash_attn_kernel<<<dim3(S_/64, B_*NH_, 2), 128, 0, stream>>>(Q16, K16, Vt16, aggw, att16);

  gemm_nt_bf16<<<gproj, 256, 0, stream>>>(att16, wo16, bout, d_out, HID_, 2*HID_, 3);
}
